// EncoderBlock_37486474559939
// MI455X (gfx1250) — matrix operand order repaired and checked by reading, not compiled, not run on hardware
//
#include <hip/hip_runtime.h>
#include <math.h>
#include <stdint.h>

typedef __attribute__((ext_vector_type(16))) _Float16 v16h;
typedef __attribute__((ext_vector_type(8)))  _Float16 v8h;
typedef __attribute__((ext_vector_type(8)))  float    v8f;

#define WMMA_F16(A, B, C) \
  __builtin_amdgcn_wmma_f32_16x16x32_f16(false, (A), false, (B), (short)0, (C), false, false)

#define S_LEN  1024
#define DMODEL 1024
#define DHEAD  64

// Async global->LDS copy (ASYNCcnt-tracked); INST_OFFSET applies to both the
// global and LDS side, so one base covers both 16B chunks.
__device__ __forceinline__ void async_copy_b256(uint32_t lds_addr, const void* gaddr) {
  asm volatile(
      "global_load_async_to_lds_b128 %0, %1, off\n\t"
      "global_load_async_to_lds_b128 %0, %1, off offset:16"
      :: "v"(lds_addr), "v"((unsigned long long)gaddr)
      : "memory");
}
__device__ __forceinline__ void async_copy_b128(uint32_t lds_addr, const void* gaddr) {
  asm volatile(
      "global_load_async_to_lds_b128 %0, %1, off"
      :: "v"(lds_addr), "v"((unsigned long long)gaddr)
      : "memory");
}
__device__ __forceinline__ void wait_async0() {
  asm volatile("s_wait_asynccnt 0x0" ::: "memory");
}

// ---------------------------------------------------------------------------
// DPP16 butterfly broadcast-reduction over each 16-lane group (pure VALU,
// replaces ds_bpermute shuffles): lane^1, lane^2, lane^7, lane^15.
// ---------------------------------------------------------------------------
template <int CTRL>
__device__ __forceinline__ float dpp_movf(float v) {
  return __builtin_bit_cast(float,
      __builtin_amdgcn_update_dpp(0, __builtin_bit_cast(int, v),
                                  CTRL, 0xf, 0xf, true));
}
__device__ __forceinline__ float rmax16(float v) {
  v = fmaxf(v, dpp_movf<0xB1>(v));   // quad_perm [1,0,3,2]
  v = fmaxf(v, dpp_movf<0x4E>(v));   // quad_perm [2,3,0,1]
  v = fmaxf(v, dpp_movf<0x141>(v));  // row_half_mirror
  v = fmaxf(v, dpp_movf<0x140>(v));  // row_mirror
  return v;
}
__device__ __forceinline__ float rsum16(float v) {
  v += dpp_movf<0xB1>(v);
  v += dpp_movf<0x4E>(v);
  v += dpp_movf<0x141>(v);
  v += dpp_movf<0x140>(v);
  return v;
}

// ---------------------------------------------------------------------------
// fp32 -> f16 elementwise conversion (weights)
// ---------------------------------------------------------------------------
__global__ __launch_bounds__(256) void cvt_f32_to_f16_kernel(
    const float* __restrict__ in, _Float16* __restrict__ out, int n) {
  int i = blockIdx.x * blockDim.x + threadIdx.x;
  if (i < n) out[i] = (_Float16)in[i];
}

// ---------------------------------------------------------------------------
// LayerNorm (unbiased std, divide by std+eps) fp32 -> f16, one row per block
// ---------------------------------------------------------------------------
__global__ __launch_bounds__(256) void ln_f16_kernel(
    const float* __restrict__ x, const float* __restrict__ alpha,
    const float* __restrict__ beta, _Float16* __restrict__ out, int D) {
  const int row = blockIdx.x;
  const float* xr = x + (size_t)row * D;
  float s = 0.f, ss = 0.f;
  for (int i = threadIdx.x; i < D; i += 256) {
    float v = xr[i];
    s += v; ss += v * v;
  }
#pragma unroll
  for (int off = 16; off >= 1; off >>= 1) {
    s  += __shfl_xor(s,  off, 32);
    ss += __shfl_xor(ss, off, 32);
  }
  __shared__ float rs[8], rss[8];
  const int w = threadIdx.x >> 5, lane = threadIdx.x & 31;
  if (lane == 0) { rs[w] = s; rss[w] = ss; }
  __syncthreads();
  s = 0.f; ss = 0.f;
#pragma unroll
  for (int i = 0; i < 8; ++i) { s += rs[i]; ss += rss[i]; }
  const float mean = s / (float)D;
  const float var  = (ss - (float)D * mean * mean) / (float)(D - 1);
  const float stdv = sqrtf(fmaxf(var, 0.f));
  const float rinv = 1.0f / (stdv + 1e-6f);
  for (int i = threadIdx.x; i < D; i += 256) {
    float v = alpha[i] * (xr[i] - mean) * rinv + beta[i];
    out[(size_t)row * D + i] = (_Float16)v;
  }
}

// ---------------------------------------------------------------------------
// Tiled WMMA GEMM: OUT[M,N] = A[M,K] @ W[N,K]^T + bias (+res) (relu?)
// Double-buffered: next tile DMA'd global->LDS (ASYNCcnt) while WMMAs run.
// Tile 128x128x32. 8 waves, each 32x64 of output.
// ---------------------------------------------------------------------------
template <bool RELU, bool RESID, bool OUT16>
__global__ __launch_bounds__(256) void gemm_wmma_kernel(
    const _Float16* __restrict__ A, const _Float16* __restrict__ W,
    const float* __restrict__ bias, const float* __restrict__ res,
    float* __restrict__ out32, _Float16* __restrict__ out16,
    int M, int N, int K) {
  __shared__ __align__(16) _Float16 Al[2][128][40];
  __shared__ __align__(16) _Float16 Wl[2][128][40];

  const int n0 = blockIdx.x * 128;
  const int m0 = blockIdx.y * 128;
  const int tid  = threadIdx.x;
  const int w    = tid >> 5;
  const int lane = tid & 31;
  const int sel  = lane >> 4;
  const int l16  = lane & 15;
  const int wm = (w >> 1) * 32;   // 0,32,64,96
  const int wn = (w & 1) * 64;    // 0,64

  const v8f z8 = {0.f, 0.f, 0.f, 0.f, 0.f, 0.f, 0.f, 0.f};
  v8f acc[2][4];
#pragma unroll
  for (int i = 0; i < 2; ++i)
#pragma unroll
    for (int j = 0; j < 4; ++j) acc[i][j] = z8;

  // per-thread 32B slice of each staged tile
  const int lrow = tid >> 1;            // 0..127
  const int lcol = (tid & 1) * 16;      // 0 or 16 (halves)
  const _Float16* ag = A + (size_t)(m0 + lrow) * K + lcol;
  const _Float16* wg = W + (size_t)(n0 + lrow) * K + lcol;
  const uint32_t ldsA[2] = {(uint32_t)(uintptr_t)&Al[0][lrow][lcol],
                            (uint32_t)(uintptr_t)&Al[1][lrow][lcol]};
  const uint32_t ldsW[2] = {(uint32_t)(uintptr_t)&Wl[0][lrow][lcol],
                            (uint32_t)(uintptr_t)&Wl[1][lrow][lcol]};

  const int nT = K >> 5;
  // prologue: stage tile 0 into buffer 0
  async_copy_b256(ldsA[0], ag);
  async_copy_b256(ldsW[0], wg);

  for (int t = 0; t < nT; ++t) {
    wait_async0();        // this wave's DMA into current buffer done
    __syncthreads();      // everyone's DMA done; prev compute done
    if (t + 1 < nT) {     // kick off next tile into the other buffer
      const int nb = (t + 1) & 1;
      async_copy_b256(ldsA[nb], ag + (t + 1) * 32);
      async_copy_b256(ldsW[nb], wg + (t + 1) * 32);
    }
    const int cur = t & 1;

    v16h a[2], bm[4];
#pragma unroll
    for (int i = 0; i < 2; ++i) {
      const int mr = wm + i * 16 + l16;
      ((v8h*)&a[i])[0] = *(const v8h*)&Al[cur][mr][sel * 8];
      ((v8h*)&a[i])[1] = *(const v8h*)&Al[cur][mr][16 + sel * 8];
    }
#pragma unroll
    for (int j = 0; j < 4; ++j) {
      const int nr = wn + j * 16 + l16;
      ((v8h*)&bm[j])[0] = *(const v8h*)&Wl[cur][nr][sel * 8];
      ((v8h*)&bm[j])[1] = *(const v8h*)&Wl[cur][nr][16 + sel * 8];
    }
#pragma unroll
    for (int i = 0; i < 2; ++i)
#pragma unroll
      for (int j = 0; j < 4; ++j)
        acc[i][j] = WMMA_F16(a[i], bm[j], acc[i][j]);

    __syncthreads();      // all reads of buffer 'cur' done before its reuse
  }

  // epilogue
#pragma unroll
  for (int i = 0; i < 2; ++i) {
#pragma unroll
    for (int j = 0; j < 4; ++j) {
      const int gn = n0 + wn + j * 16 + l16;
      const float bv = bias[gn];
#pragma unroll
      for (int r = 0; r < 8; ++r) {
        const int gm = m0 + wm + i * 16 + sel * 8 + r;
        float v = acc[i][j][r] + bv;
        if (RESID) v += res[(size_t)gm * N + gn];
        if (RELU)  v = fmaxf(v, 0.f);
        if (OUT16) out16[(size_t)gm * N + gn] = (_Float16)v;
        else       out32[(size_t)gm * N + gn] = v;
      }
    }
  }
}

// ---------------------------------------------------------------------------
// Flash attention: grid.x = B*H (128), grid.y = S/128 (8). 256 threads.
// K tile DMA'd global->LDS async; V transposed through VGPRs; softmax row
// reductions via DPP16 butterflies (no LDS traffic).
// ---------------------------------------------------------------------------
__global__ __launch_bounds__(256) void attn_flash_kernel(
    const _Float16* __restrict__ Q, const _Float16* __restrict__ K,
    const _Float16* __restrict__ V, _Float16* __restrict__ O) {
  __shared__ __align__(16) _Float16 Kt[32][72];      // [key][dh]
  __shared__ __align__(16) _Float16 Vt[64][40];      // [dh][key]
  __shared__ __align__(16) _Float16 Pl[8][16][40];   // per-wave P tile

  const int bh = blockIdx.x;
  const int b  = bh >> 4;
  const int h  = bh & 15;
  const int qbase = blockIdx.y * 128;
  const int tid  = threadIdx.x;
  const int w    = tid >> 5;
  const int lane = tid & 31;
  const int sel  = lane >> 4;
  const int l16  = lane & 15;

  const size_t headoff = (size_t)b * S_LEN * DMODEL + (size_t)h * DHEAD;

  // Q A-fragments for this wave's 16 rows (dh 0..31 and 32..63)
  const int qm = qbase + w * 16 + l16;
  v16h aq0, aq1;
  {
    const _Float16* qp = Q + headoff + (size_t)qm * DMODEL;
    ((v8h*)&aq0)[0] = *(const v8h*)(qp + sel * 8);
    ((v8h*)&aq0)[1] = *(const v8h*)(qp + 16 + sel * 8);
    ((v8h*)&aq1)[0] = *(const v8h*)(qp + 32 + sel * 8);
    ((v8h*)&aq1)[1] = *(const v8h*)(qp + 48 + sel * 8);
  }

  const v8f z8 = {0.f, 0.f, 0.f, 0.f, 0.f, 0.f, 0.f, 0.f};
  v8f o0 = z8, o1 = z8, o2 = z8, o3 = z8;
  float mrun[8], lrun[8];
#pragma unroll
  for (int r = 0; r < 8; ++r) { mrun[r] = -3.0e38f; lrun[r] = 0.f; }

  const int srow = tid >> 3;        // 0..31
  const int sc8  = (tid & 7) * 8;   // 0..56
  const uint32_t ldsK = (uint32_t)(uintptr_t)&Kt[srow][sc8];

  for (int kt = 0; kt < S_LEN; kt += 32) {
    __syncthreads();
    {  // stage K tile via async DMA; V^T via VGPR transpose
      async_copy_b128(ldsK, K + headoff + (size_t)(kt + srow) * DMODEL + sc8);
      const _Float16* vp = V + headoff + (size_t)(kt + srow) * DMODEL + sc8;
      v8h vv = *(const v8h*)vp;
#pragma unroll
      for (int j = 0; j < 8; ++j) Vt[sc8 + j][srow] = vv[j];
    }
    wait_async0();
    __syncthreads();

    // scores: two 16x16 f32 frags (keys 0..15 and 16..31 of this tile)
    v8f s0 = z8, s1 = z8;
    {
      v16h bk;
      const _Float16* kr0 = &Kt[l16][0];
      ((v8h*)&bk)[0] = *(const v8h*)(kr0 + sel * 8);
      ((v8h*)&bk)[1] = *(const v8h*)(kr0 + 16 + sel * 8);
      s0 = WMMA_F16(aq0, bk, s0);
      ((v8h*)&bk)[0] = *(const v8h*)(kr0 + 32 + sel * 8);
      ((v8h*)&bk)[1] = *(const v8h*)(kr0 + 48 + sel * 8);
      s0 = WMMA_F16(aq1, bk, s0);
      const _Float16* kr1 = &Kt[16 + l16][0];
      ((v8h*)&bk)[0] = *(const v8h*)(kr1 + sel * 8);
      ((v8h*)&bk)[1] = *(const v8h*)(kr1 + 16 + sel * 8);
      s1 = WMMA_F16(aq0, bk, s1);
      ((v8h*)&bk)[0] = *(const v8h*)(kr1 + 32 + sel * 8);
      ((v8h*)&bk)[1] = *(const v8h*)(kr1 + 48 + sel * 8);
      s1 = WMMA_F16(aq1, bk, s1);
    }

    // online softmax over this 32-key block (row = sel*8+r across 16 lanes)
    float p0a[8], p1a[8];
#pragma unroll
    for (int r = 0; r < 8; ++r) {
      float v0 = s0[r] * 0.125f;
      float v1 = s1[r] * 0.125f;
      const float mx = rmax16(fmaxf(v0, v1));
      const float mnew = fmaxf(mrun[r], mx);
      const float p0 = __expf(v0 - mnew);
      const float p1 = __expf(v1 - mnew);
      const float rsum = rsum16(p0 + p1);
      const float alpha = __expf(mrun[r] - mnew);
      mrun[r] = mnew;
      lrun[r] = lrun[r] * alpha + rsum;
      o0[r] *= alpha; o1[r] *= alpha; o2[r] *= alpha; o3[r] *= alpha;
      p0a[r] = p0; p1a[r] = p1;
    }

    // store P (C-layout) to LDS as f16, reload in A-layout
#pragma unroll
    for (int r = 0; r < 8; ++r) {
      Pl[w][sel * 8 + r][l16]      = (_Float16)p0a[r];
      Pl[w][sel * 8 + r][16 + l16] = (_Float16)p1a[r];
    }
    __syncthreads();

    v16h ap;
    ((v8h*)&ap)[0] = *(const v8h*)&Pl[w][l16][sel * 8];
    ((v8h*)&ap)[1] = *(const v8h*)&Pl[w][l16][16 + sel * 8];

    {  // O += P @ V : four 16-wide dh column blocks
      v16h bv;
      const _Float16* vr;
      vr = &Vt[l16][0];
      ((v8h*)&bv)[0] = *(const v8h*)(vr + sel * 8);
      ((v8h*)&bv)[1] = *(const v8h*)(vr + 16 + sel * 8);
      o0 = WMMA_F16(ap, bv, o0);
      vr = &Vt[16 + l16][0];
      ((v8h*)&bv)[0] = *(const v8h*)(vr + sel * 8);
      ((v8h*)&bv)[1] = *(const v8h*)(vr + 16 + sel * 8);
      o1 = WMMA_F16(ap, bv, o1);
      vr = &Vt[32 + l16][0];
      ((v8h*)&bv)[0] = *(const v8h*)(vr + sel * 8);
      ((v8h*)&bv)[1] = *(const v8h*)(vr + 16 + sel * 8);
      o2 = WMMA_F16(ap, bv, o2);
      vr = &Vt[48 + l16][0];
      ((v8h*)&bv)[0] = *(const v8h*)(vr + sel * 8);
      ((v8h*)&bv)[1] = *(const v8h*)(vr + 16 + sel * 8);
      o3 = WMMA_F16(ap, bv, o3);
    }
  }

  // finalize: divide by row sums, write f16 [B,S,H*Dh]
#pragma unroll
  for (int r = 0; r < 8; ++r) {
    const float inv = 1.0f / lrun[r];
    const size_t row = (size_t)(qbase + w * 16 + sel * 8 + r);
    _Float16* orow = O + headoff + row * DMODEL;
    orow[l16]      = (_Float16)(o0[r] * inv);
    orow[16 + l16] = (_Float16)(o1[r] * inv);
    orow[32 + l16] = (_Float16)(o2[r] * inv);
    orow[48 + l16] = (_Float16)(o3[r] * inv);
  }
}

// ---------------------------------------------------------------------------
// Host-side orchestration
// ---------------------------------------------------------------------------
extern "C" void kernel_launch(void* const* d_in, const int* in_sizes, int n_in,
                              void* d_out, int out_size, void* d_ws, size_t ws_size,
                              hipStream_t stream) {
  (void)in_sizes; (void)n_in; (void)out_size; (void)ws_size;

  const float* x   = (const float*)d_in[0];
  // d_in[1] = src_mask (unused by reference's attention)
  const float* wq  = (const float*)d_in[2];
  const float* bq  = (const float*)d_in[3];
  const float* wk  = (const float*)d_in[4];
  const float* bk  = (const float*)d_in[5];
  const float* wv  = (const float*)d_in[6];
  const float* bv  = (const float*)d_in[7];
  const float* wo  = (const float*)d_in[8];
  const float* bo  = (const float*)d_in[9];
  const float* w1  = (const float*)d_in[10];
  const float* b1  = (const float*)d_in[11];
  const float* w2  = (const float*)d_in[12];
  const float* b2  = (const float*)d_in[13];
  const float* alp = (const float*)d_in[14];
  const float* bet = (const float*)d_in[15];
  float* out = (float*)d_out;

  const int D = 1024, HF = 4096, M = 8 * 1024;

  char* ws = (char*)d_ws;
  size_t off = 0;
  auto alloc = [&](size_t bytes) -> void* {
    void* p = ws + off;
    off = (off + bytes + 255) & ~(size_t)255;
    return p;
  };

  _Float16* wq16 = (_Float16*)alloc((size_t)D * D * 2);
  _Float16* wk16 = (_Float16*)alloc((size_t)D * D * 2);
  _Float16* wv16 = (_Float16*)alloc((size_t)D * D * 2);
  _Float16* wo16 = (_Float16*)alloc((size_t)D * D * 2);
  _Float16* w116 = (_Float16*)alloc((size_t)HF * D * 2);
  _Float16* w216 = (_Float16*)alloc((size_t)D * HF * 2);
  _Float16* xn   = (_Float16*)alloc((size_t)M * D * 2);   // LN output (reused)
  _Float16* q16  = (_Float16*)alloc((size_t)M * D * 2);
  _Float16* k16  = (_Float16*)alloc((size_t)M * D * 2);
  _Float16* v16  = (_Float16*)alloc((size_t)M * D * 2);
  _Float16* at16 = (_Float16*)alloc((size_t)M * D * 2);
  _Float16* h16  = q16;  // FFN hidden (64 MB) reuses q/k/v/attn region

  const dim3 blk(256);

  // 1) convert weights fp32 -> f16
  cvt_f32_to_f16_kernel<<<(D * D + 255) / 256, blk, 0, stream>>>(wq, wq16, D * D);
  cvt_f32_to_f16_kernel<<<(D * D + 255) / 256, blk, 0, stream>>>(wk, wk16, D * D);
  cvt_f32_to_f16_kernel<<<(D * D + 255) / 256, blk, 0, stream>>>(wv, wv16, D * D);
  cvt_f32_to_f16_kernel<<<(D * D + 255) / 256, blk, 0, stream>>>(wo, wo16, D * D);
  cvt_f32_to_f16_kernel<<<(HF * D + 255) / 256, blk, 0, stream>>>(w1, w116, HF * D);
  cvt_f32_to_f16_kernel<<<(D * HF + 255) / 256, blk, 0, stream>>>(w2, w216, D * HF);

  // 2) LN1
  ln_f16_kernel<<<M, blk, 0, stream>>>(x, alp, bet, xn, D);

  // 3) Q/K/V projections
  const dim3 gD(D / 128, M / 128);   // (8, 64)
  gemm_wmma_kernel<false, false, true><<<gD, blk, 0, stream>>>(
      xn, wq16, bq, nullptr, nullptr, q16, M, D, D);
  gemm_wmma_kernel<false, false, true><<<gD, blk, 0, stream>>>(
      xn, wk16, bk, nullptr, nullptr, k16, M, D, D);
  gemm_wmma_kernel<false, false, true><<<gD, blk, 0, stream>>>(
      xn, wv16, bv, nullptr, nullptr, v16, M, D, D);

  // 4) flash attention
  attn_flash_kernel<<<dim3(128, 8), blk, 0, stream>>>(q16, k16, v16, at16);

  // 5) output projection + residual(x) -> d_out (= x1, fp32)
  gemm_wmma_kernel<false, true, false><<<gD, blk, 0, stream>>>(
      at16, wo16, bo, x, out, nullptr, M, D, D);

  // 6) LN2 on x1
  ln_f16_kernel<<<M, blk, 0, stream>>>(out, alp, bet, xn, D);

  // 7) FFN1 (ReLU) -> h16
  const dim3 gH(HF / 128, M / 128);  // (32, 64)
  gemm_wmma_kernel<true, false, true><<<gH, blk, 0, stream>>>(
      xn, w116, b1, nullptr, nullptr, h16, M, HF, D);

  // 8) FFN2 + residual(x1) -> d_out
  gemm_wmma_kernel<false, true, false><<<gD, blk, 0, stream>>>(
      h16, w216, b2, out, out, nullptr, M, D, HF);
}
